// ASNAAttention_35467839931085
// MI455X (gfx1250) — matrix operand order repaired and checked by reading, not compiled, not run on hardware
//
#include <hip/hip_runtime.h>
#include <hip/hip_bf16.h>
#include <math.h>

#define BB 4
#define NN 2048
#define DD 128
#define HH 8
#define HDIM 16
#define KK 128
#define RADIUS 0.05f
#define EPSV 1e-8f

typedef __attribute__((ext_vector_type(16))) _Float16 v16h;
typedef __attribute__((ext_vector_type(8)))  _Float16 v8h;
typedef __attribute__((ext_vector_type(8)))  float    v8f;
typedef __attribute__((ext_vector_type(4)))  int      v4i;

// async-copy builtin wants int4 pointers in explicit address spaces
#define GAS1(p) ((__attribute__((address_space(1))) v4i*)(p))
#define LAS3(p) ((__attribute__((address_space(3))) v4i*)(p))

#if __has_builtin(__builtin_amdgcn_global_load_async_to_lds_b128)
#define HAS_ASYNC_LDS 1
#else
#define HAS_ASYNC_LDS 0
#endif

__device__ __forceinline__ void wait_async0() {
#if __has_builtin(__builtin_amdgcn_s_wait_asynccnt)
    __builtin_amdgcn_s_wait_asynccnt(0);
#else
    asm volatile("s_wait_asynccnt 0x0" ::: "memory");
#endif
}

// ---------------------------------------------------------------------------
// 0a. f32 -> f16 bulk convert (features)
// ---------------------------------------------------------------------------
__global__ __launch_bounds__(256)
void f32_to_f16_kernel(const float* __restrict__ src, _Float16* __restrict__ dst) {
    int gid = (blockIdx.x * 256 + threadIdx.x) * 4;
    #pragma unroll
    for (int i = 0; i < 4; ++i) dst[gid + i] = (_Float16)src[gid + i];
}

// ---------------------------------------------------------------------------
// 0b. weight transpose + convert:  Wt[n*128+k] = (f16) W[k*128+n]
//     (column-major f16 so WMMA B-fragments are contiguous 32B runs)
// ---------------------------------------------------------------------------
__global__ __launch_bounds__(256)
void w_transpose_f16_kernel(const float* __restrict__ W, _Float16* __restrict__ Wt) {
    int gid = blockIdx.x * 256 + threadIdx.x;      // 0..16383
    int k = gid >> 7, n = gid & 127;
    Wt[n * 128 + k] = (_Float16)W[gid];
}

// ---------------------------------------------------------------------------
// 1. local density
// ---------------------------------------------------------------------------
__global__ __launch_bounds__(256)
void density_kernel(const float* __restrict__ coords, const float* __restrict__ times,
                    float* __restrict__ density) {
    int gid = blockIdx.x * 256 + threadIdx.x;
    int b = gid >> 11;
    int n = gid & (NN - 1);
    const float* cb = coords + (size_t)b * NN * 2;
    const float* tb = times  + (size_t)b * NN;
    float px = cb[n * 2], py = cb[n * 2 + 1], pt = tb[n];
    int cnt = 0;
    for (int m = 0; m < NN; ++m) {
        float dx = px - cb[m * 2];
        float dy = py - cb[m * 2 + 1];
        float dt = pt - tb[m];
        float d  = sqrtf(dx * dx + dy * dy + dt * dt + EPSV);
        cnt += (d < RADIUS) ? 1 : 0;
    }
    density[gid] = (float)cnt / ((float)NN * RADIUS * RADIUS * RADIUS + EPSV);
}

// ---------------------------------------------------------------------------
// 2. per-batch mean density
// ---------------------------------------------------------------------------
__global__ __launch_bounds__(256)
void mean_kernel(const float* __restrict__ density, float* __restrict__ meand) {
    __shared__ float red[256];
    int b = blockIdx.x;
    float s = 0.0f;
    for (int i = threadIdx.x; i < NN; i += 256) s += density[b * NN + i];
    red[threadIdx.x] = s;
    __syncthreads();
    for (int o = 128; o > 0; o >>= 1) {
        if ((int)threadIdx.x < o) red[threadIdx.x] += red[threadIdx.x + o];
        __syncthreads();
    }
    if (threadIdx.x == 0) meand[b] = red[0] / (float)NN;
}

// ---------------------------------------------------------------------------
// 3. adaptive k per point
// ---------------------------------------------------------------------------
__global__ __launch_bounds__(256)
void kvals_kernel(const float* __restrict__ density, const float* __restrict__ meand,
                  const float* __restrict__ gamma_p, int* __restrict__ kvals) {
    int gid = blockIdx.x * 256 + threadIdx.x;
    int b = gid >> 11;
    float gamma = 1.0f / (1.0f + expf(-gamma_p[0]));
    float ratio = powf(meand[b] / (density[gid] + EPSV), gamma);
    float kf = fminf(fmaxf(32.0f * ratio, 8.0f), 128.0f);
    kvals[gid] = (int)kf;
}

// ---------------------------------------------------------------------------
// 4. spatiotemporal distance + sorted top-K=128 (self excluded)
// ---------------------------------------------------------------------------
__global__ __launch_bounds__(256)
void topk_kernel(const float* __restrict__ coords, const float* __restrict__ times,
                 const float* __restrict__ swp, const float* __restrict__ twp,
                 int* __restrict__ nbr) {
    __shared__ unsigned long long keys[NN];
    __shared__ unsigned long long red[256];
    int bn = blockIdx.x;
    int b = bn >> 11;
    int n = bn & (NN - 1);
    int tid = threadIdx.x;
    float as = log1pf(expf(swp[0]));
    float at = log1pf(expf(twp[0]));
    const float* cb = coords + (size_t)b * NN * 2;
    const float* tb = times  + (size_t)b * NN;
    float px = cb[n * 2], py = cb[n * 2 + 1], pt = tb[n];
    for (int s = 0; s < NN / 256; ++s) {
        int j = tid + s * 256;
        unsigned long long key;
        if (j == n) {
            key = ~0ull;
        } else {
            float dx = px - cb[j * 2];
            float dy = py - cb[j * 2 + 1];
            float dt = pt - tb[j];
            float d  = sqrtf(as * (dx * dx + dy * dy) + at * dt * dt + EPSV);
            key = ((unsigned long long)__float_as_uint(d) << 32) | (unsigned)j;
        }
        keys[j] = key;
    }
    __syncthreads();
    for (int t = 0; t < KK; ++t) {
        unsigned long long lm = ~0ull;
        #pragma unroll
        for (int s = 0; s < NN / 256; ++s) {
            unsigned long long kv = keys[tid + s * 256];
            lm = (kv < lm) ? kv : lm;
        }
        red[tid] = lm;
        __syncthreads();
        for (int o = 128; o > 0; o >>= 1) {
            if (tid < o) {
                unsigned long long o2 = red[tid + o];
                if (o2 < red[tid]) red[tid] = o2;
            }
            __syncthreads();
        }
        if (tid == 0) {
            unsigned long long best = red[0];
            int idx = (int)(unsigned)(best & 0xffffffffull);
            nbr[(size_t)bn * KK + t] = idx;
            keys[idx] = ~0ull;
        }
        __syncthreads();
    }
}

// ---------------------------------------------------------------------------
// 5. WMMA GEMM:  Out[M,128] = A[M,128] @ W[128,128] + bias
//    A: f16 row-major.  Wt: f16 column-major (Wt[n*128+k]).
//    Both tiles staged in LDS via async global->LDS b128 copies, then each
//    wave assembles fragments with contiguous ds reads; 4 unrolled WMMAs.
// ---------------------------------------------------------------------------
__global__ __launch_bounds__(256)
void gemm128_bias_f16_kernel(const _Float16* __restrict__ Ah,
                             const _Float16* __restrict__ Wt,
                             const float* __restrict__ bias,
                             float* __restrict__ Out) {
    __shared__ __align__(16) _Float16 sA[16 * 128];     // 4 KB  A tile
    __shared__ __align__(16) _Float16 sW[128 * 128];    // 32 KB transposed weight
    const int ND = 128;
    int tid = threadIdx.x;

    const char* aSrc = (const char*)(Ah + (size_t)blockIdx.x * 16 * 128) + tid * 16;
    char* aDst = (char*)sA + tid * 16;
#if HAS_ASYNC_LDS
    __builtin_amdgcn_global_load_async_to_lds_b128(GAS1(aSrc), LAS3(aDst), 0, 0);
    #pragma unroll
    for (int i = 0; i < 8; ++i) {
        __builtin_amdgcn_global_load_async_to_lds_b128(
            GAS1((const char*)Wt + tid * 16 + i * 4096),
            LAS3((char*)sW + tid * 16 + i * 4096), 0, 0);
    }
    wait_async0();
#else
    *(uint4*)aDst = *(const uint4*)aSrc;
    #pragma unroll
    for (int i = 0; i < 8; ++i)
        *(uint4*)((char*)sW + tid * 16 + i * 4096) =
            *(const uint4*)((const char*)Wt + tid * 16 + i * 4096);
#endif
    __syncthreads();

    int wave = tid >> 5;                 // 0..7 -> column tile
    int lane = tid & 31;
    int lg   = lane >> 4;
    int ml   = lane & 15;
    int colB = wave * 16;

    v8f acc = {};
    #pragma unroll
    for (int kc = 0; kc < 4; ++kc) {
        // A fragment: halves 0-15 = K kc*32+lg*16+0..15 (row ml)
        v8h alo = *(const v8h*)(&sA[ml * 128 + kc * 32 + lg * 16]);
        v8h ahi = *(const v8h*)(&sA[ml * 128 + kc * 32 + lg * 16 + 8]);
        v16h a = __builtin_shufflevector(alo, ahi,
            0, 1, 2, 3, 4, 5, 6, 7, 8, 9, 10, 11, 12, 13, 14, 15);
        // B fragment: 16 contiguous K at column colB+ml (col-major layout)
        v16h bfr = *(const v16h*)(&sW[(colB + ml) * 128 + kc * 32 + lg * 16]);
        acc = __builtin_amdgcn_wmma_f32_16x16x32_f16(
            false, a, false, bfr, (short)0, acc, false, false);
    }
    float bv = bias[colB + ml];
    #pragma unroll
    for (int r = 0; r < 8; ++r) {        // C/D layout: M = r + 8*lg, N = ml
        int m = blockIdx.x * 16 + (lg ? 8 : 0) + r;
        Out[m * ND + colB + ml] = acc[r] + bv;
    }
}

// ---------------------------------------------------------------------------
// 6. gathered attention + rel-pos MLP bias + masked softmax + aggregate
//    one wave per (b,n,h); 4 neighbors per lane; f16 head output.
// ---------------------------------------------------------------------------
__global__ __launch_bounds__(256)
void attn_kernel(const float* __restrict__ q, const float* __restrict__ k,
                 const float* __restrict__ v,
                 const float* __restrict__ coords, const float* __restrict__ times,
                 const int* __restrict__ nbr, const int* __restrict__ kvals,
                 const float* __restrict__ W1, const float* __restrict__ b1,
                 const float* __restrict__ W2, const float* __restrict__ b2,
                 _Float16* __restrict__ outh) {
    __shared__ float sW1[192];
    __shared__ float sb1[64];
    __shared__ float sW2[512];
    __shared__ float sb2[8];
    for (int i = threadIdx.x; i < 192; i += 256) sW1[i] = W1[i];
    for (int i = threadIdx.x; i < 64;  i += 256) sb1[i] = b1[i];
    for (int i = threadIdx.x; i < 512; i += 256) sW2[i] = W2[i];
    if (threadIdx.x < 8) sb2[threadIdx.x] = b2[threadIdx.x];
    __syncthreads();

    int wave = threadIdx.x >> 5;
    int lane = threadIdx.x & 31;
    int gw = blockIdx.x * 8 + wave;              // (b*N + n)*H + h
    int h  = gw & (HH - 1);
    int bn = gw >> 3;
    int b  = bn >> 11;

    const float* qr = q + (size_t)bn * DD + h * HDIM;
    float qv[HDIM];
    #pragma unroll
    for (int d = 0; d < HDIM; ++d) qv[d] = qr[d];
    int kval = kvals[bn];
    float cx = coords[(size_t)bn * 2], cy = coords[(size_t)bn * 2 + 1];
    float ct = times[bn];

    int   idx4[4];
    float sc[4];
    float mx = -1e30f;
    #pragma unroll
    for (int t = 0; t < 4; ++t) {
        int j = lane + t * 32;
        int idx = nbr[(size_t)bn * KK + j];
        idx4[t] = idx;
        float score;
        if (j < kval) {
            size_t pi = (size_t)b * NN + idx;
            const float* kr = k + pi * DD + h * HDIM;
            __builtin_prefetch(v + pi * DD + h * HDIM, 0, 0);
            float dot = 0.0f;
            #pragma unroll
            for (int d = 0; d < HDIM; ++d) dot += qv[d] * kr[d];
            float r0 = coords[pi * 2]     - cx;
            float r1 = coords[pi * 2 + 1] - cy;
            float r2 = times[pi]          - ct;
            float acc = 0.0f;
            for (int c = 0; c < 64; ++c) {
                float hh = r0 * sW1[c] + r1 * sW1[64 + c] + r2 * sW1[128 + c] + sb1[c];
                float g  = 0.5f * hh * (1.0f + erff(hh * 0.70710678118f));
                acc += g * sW2[c * HH + h];
            }
            score = dot * 0.25f + acc + sb2[h];
        } else {
            score = -1e30f;
        }
        sc[t] = score;
        mx = fmaxf(mx, score);
    }
    #pragma unroll
    for (int o = 16; o > 0; o >>= 1) mx = fmaxf(mx, __shfl_xor(mx, o));
    float p[4];
    float s = 0.0f;
    #pragma unroll
    for (int t = 0; t < 4; ++t) { p[t] = expf(sc[t] - mx); s += p[t]; }
    #pragma unroll
    for (int o = 16; o > 0; o >>= 1) s += __shfl_xor(s, o);
    float inv = 1.0f / s;

    float facc[HDIM];
    #pragma unroll
    for (int d = 0; d < HDIM; ++d) facc[d] = 0.0f;
    #pragma unroll
    for (int t = 0; t < 4; ++t) {
        float pr = p[t] * inv;
        const float* vr = v + ((size_t)b * NN + idx4[t]) * DD + h * HDIM;
        #pragma unroll
        for (int d = 0; d < HDIM; ++d) facc[d] += pr * vr[d];
    }
    #pragma unroll
    for (int d = 0; d < HDIM; ++d) {
        #pragma unroll
        for (int o = 16; o > 0; o >>= 1) facc[d] += __shfl_xor(facc[d], o);
    }
    if (lane == 0) {
        _Float16* orow = outh + (size_t)bn * DD + h * HDIM;
        #pragma unroll
        for (int d = 0; d < HDIM; ++d) orow[d] = (_Float16)facc[d];
    }
}

// ---------------------------------------------------------------------------
extern "C" void kernel_launch(void* const* d_in, const int* in_sizes, int n_in,
                              void* d_out, int out_size, void* d_ws, size_t ws_size,
                              hipStream_t stream) {
    const float* features = (const float*)d_in[0];
    const float* coords   = (const float*)d_in[1];
    const float* times    = (const float*)d_in[2];
    const float* Wq = (const float*)d_in[3];  const float* bq = (const float*)d_in[4];
    const float* Wk = (const float*)d_in[5];  const float* bk = (const float*)d_in[6];
    const float* Wv = (const float*)d_in[7];  const float* bv = (const float*)d_in[8];
    const float* Wo = (const float*)d_in[9];  const float* bo = (const float*)d_in[10];
    const float* spw = (const float*)d_in[11];
    const float* tpw = (const float*)d_in[12];
    const float* gmp = (const float*)d_in[13];
    const float* W1 = (const float*)d_in[14]; const float* b1 = (const float*)d_in[15];
    const float* W2 = (const float*)d_in[16]; const float* b2 = (const float*)d_in[17];

    char* ws = (char*)d_ws;
    auto carve = [&](size_t bytes) -> void* {
        void* p = (void*)ws;
        ws += (bytes + 255) & ~(size_t)255;
        return p;
    };
    const size_t MTOT = (size_t)BB * NN;            // 8192 rows
    float*     density = (float*)carve(MTOT * sizeof(float));
    float*     meand   = (float*)carve((size_t)BB * sizeof(float));
    int*       kvals   = (int*)  carve(MTOT * sizeof(int));
    int*       nbr     = (int*)  carve(MTOT * KK * sizeof(int));
    _Float16*  feat_h  = (_Float16*)carve(MTOT * DD * sizeof(_Float16));
    _Float16*  wqT     = (_Float16*)carve((size_t)DD * DD * sizeof(_Float16));
    _Float16*  wkT     = (_Float16*)carve((size_t)DD * DD * sizeof(_Float16));
    _Float16*  wvT     = (_Float16*)carve((size_t)DD * DD * sizeof(_Float16));
    _Float16*  woT     = (_Float16*)carve((size_t)DD * DD * sizeof(_Float16));
    float*     qb      = (float*)carve(MTOT * DD * sizeof(float));
    float*     kb      = (float*)carve(MTOT * DD * sizeof(float));
    float*     vb      = (float*)carve(MTOT * DD * sizeof(float));
    _Float16*  ohh     = (_Float16*)carve(MTOT * DD * sizeof(_Float16));

    // precision prep: f16 activations + transposed f16 weights for WMMA
    f32_to_f16_kernel<<<(int)(MTOT * DD / 1024), 256, 0, stream>>>(features, feat_h);
    w_transpose_f16_kernel<<<64, 256, 0, stream>>>(Wq, wqT);
    w_transpose_f16_kernel<<<64, 256, 0, stream>>>(Wk, wkT);
    w_transpose_f16_kernel<<<64, 256, 0, stream>>>(Wv, wvT);
    w_transpose_f16_kernel<<<64, 256, 0, stream>>>(Wo, woT);

    density_kernel<<<(int)(MTOT / 256), 256, 0, stream>>>(coords, times, density);
    mean_kernel<<<BB, 256, 0, stream>>>(density, meand);
    kvals_kernel<<<(int)(MTOT / 256), 256, 0, stream>>>(density, meand, gmp, kvals);
    topk_kernel<<<(int)MTOT, 256, 0, stream>>>(coords, times, spw, tpw, nbr);

    gemm128_bias_f16_kernel<<<(int)(MTOT / 16), 256, 0, stream>>>(feat_h, wqT, bq, qb);
    gemm128_bias_f16_kernel<<<(int)(MTOT / 16), 256, 0, stream>>>(feat_h, wkT, bk, kb);
    gemm128_bias_f16_kernel<<<(int)(MTOT / 16), 256, 0, stream>>>(feat_h, wvT, bv, vb);

    attn_kernel<<<(int)(MTOT * HH / 8), 256, 0, stream>>>(qb, kb, vb, coords, times,
                                                          nbr, kvals, W1, b1, W2, b2, ohh);

    gemm128_bias_f16_kernel<<<(int)(MTOT / 16), 256, 0, stream>>>(ohh, woT, bo, (float*)d_out);
}
